// RNN_84035330113984
// MI455X (gfx1250) — matrix operand order repaired and checked by reading, not compiled, not run on hardware
//
#include <hip/hip_runtime.h>

// ---------------------------------------------------------------------------
// Grid-cell Elman RNN on MI455X (gfx1250).
//   h0 = p0 @ W_enc^T ; h_{t+1} = tanh(v_t @ W_ih^T + h_t @ W_hh^T)
//   out_t = h_{t+1} @ W_dec^T
//
// bf16 WMMA (v_wmma_f32_16x16x32_bf16, fp32 accumulate) for all GEMMs.
// LDS-tiled: block = 256 threads (8 waves) computes a 64(M) x 128(N) tile;
// each wave owns a 32x32 sub-tile (2x2 accumulators). K staged through LDS
// in chunks of 128. Software-pipelined: the global fetch of chunk k+1 is in
// flight during the 16-WMMA compute phase of chunk k, so HBM/L2 latency is
// hidden behind matrix math. LDS rows padded to 136 bf16 (68 DWORDs ->
// bank step 4) for conflict-free 16-B fragment reads.
// One kernel launch per timestep handles the serial recurrence.
// ---------------------------------------------------------------------------

typedef __attribute__((ext_vector_type(16))) __bf16 v16bf;
typedef __attribute__((ext_vector_type(8)))  __bf16 v8bf;
typedef __attribute__((ext_vector_type(8)))  float  v8f;

#define RNN_T   100
#define RNN_B   256
#define RNN_NG  2048
#define RNN_NP  512

#define M_BLK   64
#define N_BLK   128
#define K_BLK   128
#define THREADS 256
#define LDS_PITCH 136   // bf16 elems per LDS row (128 data + 8 pad), 272 B

#define MODE_STORE_BF16      0   // encoder: store bf16 h0
#define MODE_TANH_STORE_BF16 1   // recurrent step: +v@W_ih^T, tanh, store bf16
#define MODE_STORE_F32       2   // decoder: store fp32 predictions

// ---------------------------------------------------------------------------
// fp32 -> bf16 conversion (weights / p0 staged once into workspace)
// ---------------------------------------------------------------------------
__global__ void f32_to_bf16_kernel(const float* __restrict__ src,
                                   __bf16* __restrict__ dst, int n) {
  int i = blockIdx.x * blockDim.x + threadIdx.x;
  if (i < n) dst[i] = (__bf16)src[i];
}

// ---------------------------------------------------------------------------
// C[M,N] = A[M,K] @ W[N,K]^T, LDS-tiled WMMA. Epilogue selected by `mode`.
//
// Fragment layouts (CDNA5 ISA, 16-bit, wave32):
//   A 16x32: lanes 0-15 -> row M=lane,    K {k..k+7,  k+16..k+23}
//            lanes16-31 -> row M=lane-16, K {k+8..15, k+24..31}
//   B 32x16: lanes 0-15 -> col N=lane,    K {k..k+7,  k+16..k+23}
//            lanes16-31 -> col N=lane-16, K {k+8..15, k+24..31}
//   C/D 16x16: vgpr r of lane L -> (M = r + 8*(L>>4), N = L&15)
// ---------------------------------------------------------------------------
__global__ __launch_bounds__(THREADS)
void wmma_gemm_bt_kernel(const __bf16* __restrict__ A,   // [M,K] row-major
                         const __bf16* __restrict__ W,   // [N,K] row-major
                         void* __restrict__ outp,        // [M,N]
                         int M, int N, int K,
                         const float* __restrict__ vt,   // [M,2]  (mode 1)
                         const float* __restrict__ wih,  // [N,2]  (mode 1)
                         int mode) {
  __shared__ __bf16 lA[M_BLK * LDS_PITCH];  // 17.0 KB
  __shared__ __bf16 lB[N_BLK * LDS_PITCH];  // 34.0 KB

  const int tid  = threadIdx.x;
  const int lane = tid & 31;
  const int wave = tid >> 5;
  const int wm   = wave >> 2;   // 0..1 : wave row in 2x4 wave grid
  const int wn   = wave & 3;    // 0..3 : wave col

  const int nblk   = N / N_BLK;
  const int bm     = blockIdx.x / nblk;
  const int bn     = blockIdx.x - bm * nblk;
  const int m_base = bm * M_BLK;
  const int n_base = bn * N_BLK;

  const int hi = lane >> 4;     // half-wave select
  const int lo = lane & 15;

  // ---- global -> LDS staging map (per K_BLK chunk) ----
  // A tile: 64 rows x 128 elems (16 KB), 4 threads/row, 64 B per thread.
  // B tile: 128 rows x 128 elems (32 KB), 2 threads/row, 128 B per thread.
  const int arow = tid >> 2;
  const int acol = (tid & 3) * 32;
  const int brow = tid >> 1;
  const int bcol = (tid & 1) * 64;

  const __bf16* Ag  = A + (size_t)(m_base + arow) * K + acol;
  const __bf16* Wg  = W + (size_t)(n_base + brow) * K + bcol;
  __bf16* lAw = lA + arow * LDS_PITCH + acol;
  __bf16* lBw = lB + brow * LDS_PITCH + bcol;

  v8f acc[2][2] = {};

  // ---- prologue: fetch chunk 0 into registers ----
  v8bf a0 = *(const v8bf*)(Ag + 0);
  v8bf a1 = *(const v8bf*)(Ag + 8);
  v8bf a2 = *(const v8bf*)(Ag + 16);
  v8bf a3 = *(const v8bf*)(Ag + 24);
  v8bf b0 = *(const v8bf*)(Wg + 0);
  v8bf b1 = *(const v8bf*)(Wg + 8);
  v8bf b2 = *(const v8bf*)(Wg + 16);
  v8bf b3 = *(const v8bf*)(Wg + 24);
  v8bf b4 = *(const v8bf*)(Wg + 32);
  v8bf b5 = *(const v8bf*)(Wg + 40);
  v8bf b6 = *(const v8bf*)(Wg + 48);
  v8bf b7 = *(const v8bf*)(Wg + 56);
  __builtin_prefetch((const void*)(Ag + K_BLK), 0, 1);
  __builtin_prefetch((const void*)(Wg + K_BLK), 0, 1);

  for (int k0 = 0; k0 < K; k0 += K_BLK) {
    __syncthreads();  // previous round's LDS reads complete (WAR)
    *(v8bf*)(lAw +  0) = a0;  *(v8bf*)(lAw +  8) = a1;
    *(v8bf*)(lAw + 16) = a2;  *(v8bf*)(lAw + 24) = a3;
    *(v8bf*)(lBw +  0) = b0;  *(v8bf*)(lBw +  8) = b1;
    *(v8bf*)(lBw + 16) = b2;  *(v8bf*)(lBw + 24) = b3;
    *(v8bf*)(lBw + 32) = b4;  *(v8bf*)(lBw + 40) = b5;
    *(v8bf*)(lBw + 48) = b6;  *(v8bf*)(lBw + 56) = b7;
    __syncthreads();  // tile visible to all waves

    // Issue global loads of chunk k+1 now; they stay in flight behind the
    // 16 WMMAs below and are only waited on at the next iteration's stores.
    const int kn = k0 + K_BLK;
    if (kn < K) {
      a0 = *(const v8bf*)(Ag + kn);
      a1 = *(const v8bf*)(Ag + kn + 8);
      a2 = *(const v8bf*)(Ag + kn + 16);
      a3 = *(const v8bf*)(Ag + kn + 24);
      b0 = *(const v8bf*)(Wg + kn);
      b1 = *(const v8bf*)(Wg + kn + 8);
      b2 = *(const v8bf*)(Wg + kn + 16);
      b3 = *(const v8bf*)(Wg + kn + 24);
      b4 = *(const v8bf*)(Wg + kn + 32);
      b5 = *(const v8bf*)(Wg + kn + 40);
      b6 = *(const v8bf*)(Wg + kn + 48);
      b7 = *(const v8bf*)(Wg + kn + 56);
      __builtin_prefetch((const void*)(Ag + kn + K_BLK), 0, 1);
      __builtin_prefetch((const void*)(Wg + kn + K_BLK), 0, 1);
    }

#pragma unroll
    for (int kk = 0; kk < K_BLK; kk += 32) {
      v16bf afrag[2], bfrag[2];
#pragma unroll
      for (int i = 0; i < 2; ++i) {
        const __bf16* pA = lA + (wm * 32 + i * 16 + lo) * LDS_PITCH
                              + kk + (hi ? 8 : 0);
        v8bf x0 = *(const v8bf*)pA;
        v8bf x1 = *(const v8bf*)(pA + 16);
        afrag[i] = __builtin_shufflevector(x0, x1,
                                           0, 1, 2, 3, 4, 5, 6, 7,
                                           8, 9, 10, 11, 12, 13, 14, 15);
      }
#pragma unroll
      for (int j = 0; j < 2; ++j) {
        const __bf16* pB = lB + (wn * 32 + j * 16 + lo) * LDS_PITCH
                              + kk + (hi ? 8 : 0);
        v8bf y0 = *(const v8bf*)pB;
        v8bf y1 = *(const v8bf*)(pB + 16);
        bfrag[j] = __builtin_shufflevector(y0, y1,
                                           0, 1, 2, 3, 4, 5, 6, 7,
                                           8, 9, 10, 11, 12, 13, 14, 15);
      }
#pragma unroll
      for (int i = 0; i < 2; ++i)
#pragma unroll
        for (int j = 0; j < 2; ++j)
          acc[i][j] = __builtin_amdgcn_wmma_f32_16x16x32_bf16(
              false, afrag[i], false, bfrag[j],
              (short)0, acc[i][j], false, false);
    }
  }

  // ---- epilogue ----
#pragma unroll
  for (int i = 0; i < 2; ++i) {
#pragma unroll
    for (int j = 0; j < 2; ++j) {
      const int m0 = m_base + wm * 32 + i * 16;
      const int n0 = n_base + wn * 32 + j * 16;
      const int nn = n0 + lo;
      if (mode == MODE_STORE_F32) {
        float* out = (float*)outp;
#pragma unroll
        for (int r = 0; r < 8; ++r) {
          int mm = m0 + r + (hi << 3);
          out[(size_t)mm * N + nn] = acc[i][j][r];
        }
      } else if (mode == MODE_TANH_STORE_BF16) {
        __bf16* out = (__bf16*)outp;
        const float w0 = wih[nn * 2 + 0];
        const float w1 = wih[nn * 2 + 1];
#pragma unroll
        for (int r = 0; r < 8; ++r) {
          int mm = m0 + r + (hi << 3);
          float x = acc[i][j][r] + vt[mm * 2 + 0] * w0 + vt[mm * 2 + 1] * w1;
          out[(size_t)mm * N + nn] = (__bf16)tanhf(x);
        }
      } else {  // MODE_STORE_BF16
        __bf16* out = (__bf16*)outp;
#pragma unroll
        for (int r = 0; r < 8; ++r) {
          int mm = m0 + r + (hi << 3);
          out[(size_t)mm * N + nn] = (__bf16)acc[i][j][r];
        }
      }
    }
  }
}

// ---------------------------------------------------------------------------
// Host-side orchestration (graph-capturable: only async launches on stream).
// ---------------------------------------------------------------------------
extern "C" void kernel_launch(void* const* d_in, const int* in_sizes, int n_in,
                              void* d_out, int out_size, void* d_ws, size_t ws_size,
                              hipStream_t stream) {
  const float* v     = (const float*)d_in[0]; // [T,B,2]
  const float* p0    = (const float*)d_in[1]; // [B,NP]
  const float* W_enc = (const float*)d_in[2]; // [NG,NP]
  const float* W_ih  = (const float*)d_in[3]; // [NG,2]
  const float* W_hh  = (const float*)d_in[4]; // [NG,NG]
  const float* W_dec = (const float*)d_in[5]; // [NP,NG]
  float* out = (float*)d_out;                 // [T,B,NP]

  // Workspace layout (bf16 buffers, 256B aligned). ~14.5 MB total.
  char* ws = (char*)d_ws;
  size_t off = 0;
  auto take = [&](size_t elems) -> __bf16* {
    __bf16* p = (__bf16*)(ws + off);
    off = (off + elems * sizeof(__bf16) + 255) & ~(size_t)255;
    return p;
  };
  __bf16* Wenc_b = take((size_t)RNN_NG * RNN_NP);
  __bf16* Whh_b  = take((size_t)RNN_NG * RNN_NG);
  __bf16* Wdec_b = take((size_t)RNN_NP * RNN_NG);
  __bf16* p0_b   = take((size_t)RNN_B  * RNN_NP);
  __bf16* hbuf0  = take((size_t)RNN_B  * RNN_NG);
  __bf16* hbuf1  = take((size_t)RNN_B  * RNN_NG);
  __bf16* hbuf[2] = {hbuf0, hbuf1};

  auto conv = [&](const float* s, __bf16* d, int n) {
    f32_to_bf16_kernel<<<(n + 255) / 256, 256, 0, stream>>>(s, d, n);
  };
  conv(W_enc, Wenc_b, RNN_NG * RNN_NP);
  conv(W_hh,  Whh_b,  RNN_NG * RNN_NG);
  conv(W_dec, Wdec_b, RNN_NP * RNN_NG);
  conv(p0,    p0_b,   RNN_B  * RNN_NP);

  // Encoder: h0 = p0 @ W_enc^T   [256 x 2048], K=512 -> 4x16 = 64 blocks.
  wmma_gemm_bt_kernel<<<(RNN_B / M_BLK) * (RNN_NG / N_BLK), THREADS, 0, stream>>>(
      p0_b, Wenc_b, hbuf[0], RNN_B, RNN_NG, RNN_NP,
      nullptr, nullptr, MODE_STORE_BF16);

  // Sequential recurrence + per-step decoder.
  for (int t = 0; t < RNN_T; ++t) {
    const __bf16* hin  = hbuf[t & 1];
    __bf16*       hout = hbuf[(t + 1) & 1];

    // h_{t+1} = tanh(v_t @ W_ih^T + h_t @ W_hh^T)   [256 x 2048], K=2048
    wmma_gemm_bt_kernel<<<(RNN_B / M_BLK) * (RNN_NG / N_BLK), THREADS, 0, stream>>>(
        hin, Whh_b, hout, RNN_B, RNN_NG, RNN_NG,
        v + (size_t)t * RNN_B * 2, W_ih, MODE_TANH_STORE_BF16);

    // out_t = h_{t+1} @ W_dec^T   [256 x 512], K=2048 -> 4x4 = 16 blocks.
    wmma_gemm_bt_kernel<<<(RNN_B / M_BLK) * (RNN_NP / N_BLK), THREADS, 0, stream>>>(
        hout, Wdec_b, out + (size_t)t * RNN_B * RNN_NP,
        RNN_B, RNN_NP, RNN_NG, nullptr, nullptr, MODE_STORE_F32);
  }
}
